// Seq2Seq_61177514164505
// MI455X (gfx1250) — matrix operand order repaired and checked by reading, not compiled, not run on hardware
//
#include <hip/hip_runtime.h>
#include <math.h>

// ---------------------------------------------------------------------------
// Types for gfx1250 WMMA (wave32): D(16x16,f32) = A(16x32,bf16) x B(32x16,bf16) + C
// ---------------------------------------------------------------------------
typedef __bf16 bf16_t;
typedef __bf16 v16bf __attribute__((ext_vector_type(16)));
typedef float  v8f   __attribute__((ext_vector_type(8)));

union FragU { v16bf v; uint4 u[2]; };

static __device__ __forceinline__ v8f vzero8() {
  v8f z;
#pragma unroll
  for (int i = 0; i < 8; ++i) z[i] = 0.0f;
  return z;
}

static __device__ __forceinline__ v8f wmma_bf16(v16bf a, v16bf b, v8f c) {
  // (neg_a, A, neg_b, B, c_mod, C, reuse_a, reuse_b)
  return __builtin_amdgcn_wmma_f32_16x16x32_bf16(false, a, false, b, (short)0, c,
                                                 false, false);
}

// A fragment: A row-major [M x K] bf16. Lane l holds row m0+(l&15).
// VGPR v holds K pairs: v<4 -> K = 2v + 8*(l>>4); v>=4 -> K = 2v+8 + 8*(l>>4).
// => two contiguous 128-bit loads at element offsets (8*(l>>4)) and (8*(l>>4)+16).
static __device__ __forceinline__ v16bf fragA(const bf16_t* A, int lda, int m0,
                                              int k0, int lane) {
  int m  = m0 + (lane & 15);
  int kh = (lane >> 4) * 8;
  const bf16_t* p = A + (size_t)m * lda + k0 + kh;
  FragU f;
  f.u[0] = *(const uint4*)(p);
  f.u[1] = *(const uint4*)(p + 16);
  return f.v;
}

// B fragment from pre-transposed Bt row-major [N x K] bf16 (== B column-major).
// Lane l holds col n0+(l&15); needs K = k0 + 8*(l>>4) + (0..7) and k0 + 16 + 8*(l>>4) + (0..7).
static __device__ __forceinline__ v16bf fragB(const bf16_t* Bt, int ldb, int n0,
                                              int k0, int lane) {
  int n  = n0 + (lane & 15);
  int kk = k0 + (lane >> 4) * 8;
  const bf16_t* p = Bt + (size_t)n * ldb + kk;
  FragU f;
  f.u[0] = *(const uint4*)(p);
  f.u[1] = *(const uint4*)(p + 16);
  return f.v;
}

static __device__ __forceinline__ float sigf(float x) {
  return 1.0f / (1.0f + __expf(-x));
}

// ---------------------------------------------------------------------------
// Problem dimensions (match reference)
// ---------------------------------------------------------------------------
#define BB   32
#define LL   400
#define TT   64
#define EE   256
#define HH   512
#define H2   1024
#define VV   32000
#define OOVN 100
#define VEXT (VV + OOVN)

// ---------------------------------------------------------------------------
// WMMA GEMM: C[M,N] = A[M,K](bf16,row) * Bt[N,K]^T(bf16) (+ bias[N])
// One wave per block, 32(M) x 32(N) tile so each B fragment feeds TWO WMMAs:
// the weight stream (dominant traffic) is read exactly once for the whole M=32
// batch. Software-pipelined K loop: next k-block's fragments load while the
// current four WMMAs execute (the s_wait_loadcnt moves to the loop head).
// Requires M%32==0, N%32==0, K%32==0 (all our shapes satisfy this).
// ---------------------------------------------------------------------------
__global__ __launch_bounds__(32) void s2s_gemm_wmma(
    float* __restrict__ C, const bf16_t* __restrict__ A,
    const bf16_t* __restrict__ Bt, const float* __restrict__ bias,
    int M, int N, int K) {
  const int lane = threadIdx.x;
  const int m0 = blockIdx.x * 32;
  const int n0 = blockIdx.y * 32;
  v8f acc00 = vzero8(), acc10 = vzero8(), acc01 = vzero8(), acc11 = vzero8();

  // prologue: k-block 0
  v16bf a0 = fragA(A, K, m0, 0, lane);
  v16bf a1 = fragA(A, K, m0 + 16, 0, lane);
  v16bf b0 = fragB(Bt, K, n0, 0, lane);
  v16bf b1 = fragB(Bt, K, n0 + 16, 0, lane);

  const int kblocks = K >> 5;
  for (int kb = 1; kb < kblocks; ++kb) {
    const int kn = kb << 5;
    // prefetch the weight stream one k-block further ahead (global_prefetch_b8)
    __builtin_prefetch((const void*)(Bt + (size_t)(n0 + (lane & 15)) * K + kn + 32), 0, 1);
    v16bf a0n = fragA(A, K, m0, kn, lane);
    v16bf a1n = fragA(A, K, m0 + 16, kn, lane);
    v16bf b0n = fragB(Bt, K, n0, kn, lane);
    v16bf b1n = fragB(Bt, K, n0 + 16, kn, lane);
    acc00 = wmma_bf16(a0, b0, acc00);
    acc10 = wmma_bf16(a1, b0, acc10);
    acc01 = wmma_bf16(a0, b1, acc01);
    acc11 = wmma_bf16(a1, b1, acc11);
    a0 = a0n; a1 = a1n; b0 = b0n; b1 = b1n;
  }
  acc00 = wmma_bf16(a0, b0, acc00);
  acc10 = wmma_bf16(a1, b0, acc10);
  acc01 = wmma_bf16(a0, b1, acc01);
  acc11 = wmma_bf16(a1, b1, acc11);

  const int hi = lane >> 4, lo = lane & 15;
#pragma unroll
  for (int nt = 0; nt < 2; ++nt) {
    int n = n0 + nt * 16 + lo;
    float bb = bias ? bias[n] : 0.0f;
    v8f c0 = nt ? acc01 : acc00;
    v8f c1 = nt ? acc11 : acc10;
#pragma unroll
    for (int e = 0; e < 8; ++e) {
      int m = m0 + e + 8 * hi;  // C/D layout: VGPR e -> row e (lanes 0-15), e+8 (16-31)
      C[(size_t)m * N + n] = c0[e] + bb;
      C[(size_t)(m + 16) * N + n] = c1[e] + bb;
    }
  }
}

// ---------------------------------------------------------------------------
// Weight convert+transpose: Wt[n*K+k] = (bf16) W[k*N+n]
// ---------------------------------------------------------------------------
__global__ void s2s_transpose_bf16(bf16_t* __restrict__ Wt,
                                   const float* __restrict__ W, int K, int N) {
  size_t i = (size_t)blockIdx.x * blockDim.x + threadIdx.x;
  size_t tot = (size_t)K * N;
  if (i < tot) {
    int n = (int)(i / K);
    int k = (int)(i % K);
    Wt[i] = (bf16_t)W[(size_t)k * N + n];
  }
}

// Encoder embedding gather -> xe stored [L][B][E] bf16 (so LSTM step rows are contiguous)
__global__ void s2s_embed_enc(bf16_t* __restrict__ xe, const int* __restrict__ x,
                              const float* __restrict__ embedding) {
  size_t i = (size_t)blockIdx.x * blockDim.x + threadIdx.x;
  if (i < (size_t)LL * BB * EE) {
    int l = (int)(i / (BB * EE));
    int r = (int)(i % (BB * EE));
    int b = r >> 8;  // E = 256
    int e = r & 255;
    int tok = x[b * LL + l];
    xe[i] = (bf16_t)embedding[(size_t)tok * EE + e];
  }
}

// ---------------------------------------------------------------------------
// Persistent bidirectional LSTM: one block per direction, 16 waves.
// pre = xe@Wi + b precomputed, layout [L][B][4H] fp32.
// Each wave owns hidden slice j in [w*32, w*32+32) across all 4 gates, so the
// gate nonlinearity and cell state c stay wave-local (c lives in VGPRs in the
// WMMA D layout). h is double-buffered in LDS as bf16 for the next step's A.
// B-fragments of a k-block are batch-loaded (one clause / one wait) before the
// 16-WMMA burst.
// ---------------------------------------------------------------------------
#define HPAD 520
__global__ __launch_bounds__(512) void s2s_lstm_seq(
    float* __restrict__ enc_out,          // [B][L][2H] fp32 (fwd writes 0:512, bwd 512:1024)
    const float* __restrict__ pre_f,      // [L][B][2048]
    const float* __restrict__ pre_b,
    const bf16_t* __restrict__ Wh_f_t,    // [2048][512] bf16 (transposed Wh)
    const bf16_t* __restrict__ Wh_b_t) {
  const int dir = blockIdx.x;
  const float* pre = dir ? pre_b : pre_f;
  const bf16_t* Wh = dir ? Wh_b_t : Wh_f_t;

  const int tid = threadIdx.x;
  const int lane = tid & 31;
  const int wave = tid >> 5;          // 0..15, owns j in [wave*32, wave*32+32)
  const int hi = lane >> 4, lo = lane & 15;

  __shared__ bf16_t hsh[2][32 * HPAD];
  for (int i = tid; i < 32 * HPAD; i += 512) {
    hsh[0][i] = (bf16_t)0.0f;
    hsh[1][i] = (bf16_t)0.0f;
  }
  __syncthreads();

  float cst[2][2][8];
#pragma unroll
  for (int mt = 0; mt < 2; ++mt)
#pragma unroll
    for (int nt = 0; nt < 2; ++nt)
#pragma unroll
      for (int e = 0; e < 8; ++e) cst[mt][nt][e] = 0.0f;

  int pb = 0;
  for (int it = 0; it < LL; ++it) {
    const int t = dir ? (LL - 1 - it) : it;

    v8f acc[4][2][2];
#pragma unroll
    for (int g = 0; g < 4; ++g)
#pragma unroll
      for (int mt = 0; mt < 2; ++mt)
#pragma unroll
        for (int nt = 0; nt < 2; ++nt) acc[g][mt][nt] = vzero8();

    const bf16_t* hbuf = hsh[pb];
    for (int k0 = 0; k0 < HH; k0 += 32) {
      // batch all 8 weight fragments for this k-block (single clause + wait)
      v16bf bfr[8];
#pragma unroll
      for (int g = 0; g < 4; ++g)
#pragma unroll
        for (int nt = 0; nt < 2; ++nt)
          bfr[g * 2 + nt] = fragB(Wh, HH, g * HH + wave * 32 + nt * 16, k0, lane);
      v16bf a0 = fragA(hbuf, HPAD, 0, k0, lane);
      v16bf a1 = fragA(hbuf, HPAD, 16, k0, lane);
#pragma unroll
      for (int g = 0; g < 4; ++g) {
#pragma unroll
        for (int nt = 0; nt < 2; ++nt) {
          acc[g][0][nt] = wmma_bf16(a0, bfr[g * 2 + nt], acc[g][0][nt]);
          acc[g][1][nt] = wmma_bf16(a1, bfr[g * 2 + nt], acc[g][1][nt]);
        }
      }
    }

    // Gate math + state update + write h into the other LDS buffer.
    const float* prow = pre + (size_t)t * BB * (4 * HH);
    bf16_t* hout = hsh[pb ^ 1];
#pragma unroll
    for (int mt = 0; mt < 2; ++mt) {
#pragma unroll
      for (int nt = 0; nt < 2; ++nt) {
#pragma unroll
        for (int e = 0; e < 8; ++e) {
          int m = e + 8 * hi + 16 * mt;              // batch row 0..31
          int j = wave * 32 + nt * 16 + lo;          // hidden index 0..511
          const float* pz = prow + (size_t)m * (4 * HH);
          float iv = acc[0][mt][nt][e] + pz[j];
          float fv = acc[1][mt][nt][e] + pz[HH + j];
          float gv = acc[2][mt][nt][e] + pz[2 * HH + j];
          float ov = acc[3][mt][nt][e] + pz[3 * HH + j];
          float c = sigf(fv) * cst[mt][nt][e] + sigf(iv) * tanhf(gv);
          cst[mt][nt][e] = c;
          float h = sigf(ov) * tanhf(c);
          hout[m * HPAD + j] = (bf16_t)h;
          enc_out[((size_t)m * LL + t) * (2 * HH) + dir * HH + j] = h;
        }
      }
    }
    __syncthreads();
    pb ^= 1;
  }
}

// ---------------------------------------------------------------------------
// Decoder init: dec_in = START, dec_h0 = concat(hN_f, hN_b), acc = 0
// ---------------------------------------------------------------------------
__global__ void s2s_dec_init(int* __restrict__ dec_in, float* __restrict__ dec_h,
                             bf16_t* __restrict__ dec_h_bf, float* __restrict__ accb,
                             const float* __restrict__ enc_out) {
  int i = blockIdx.x * blockDim.x + threadIdx.x;
  if (i < BB) dec_in[i] = 2;  // START
  if (i < BB * LL) accb[i] = 0.0f;
  if (i < BB * H2) {
    int b = i >> 10, k = i & 1023;
    float v = (k < HH) ? enc_out[((size_t)b * LL + (LL - 1)) * H2 + k]   // hN_f
                       : enc_out[((size_t)b * LL + 0) * H2 + k];         // hN_b
    dec_h[i] = v;
    dec_h_bf[i] = (bf16_t)v;
  }
}

__global__ void s2s_embed_dec(bf16_t* __restrict__ emb, const int* __restrict__ dec_in,
                              const float* __restrict__ embedding) {
  int i = blockIdx.x * blockDim.x + threadIdx.x;
  if (i < BB * EE) {
    int b = i >> 8, e = i & 255;
    emb[i] = (bf16_t)embedding[(size_t)dec_in[b] * EE + e];
  }
}

// GRU combine: r/z/n coupling elementwise per hidden index, then h update.
__global__ void s2s_gru_combine(float* __restrict__ h, bf16_t* __restrict__ h_bf,
                                float* __restrict__ prebuf, const float* __restrict__ zi,
                                const float* __restrict__ zh, int t) {
  int i = blockIdx.x * blockDim.x + threadIdx.x;
  if (i >= BB * H2) return;
  int b = i >> 10, j = i & 1023;
  const float* zib = zi + (size_t)b * 3 * H2;
  const float* zhb = zh + (size_t)b * 3 * H2;
  float r = sigf(zib[j] + zhb[j]);
  float z = sigf(zib[H2 + j] + zhb[H2 + j]);
  float n = tanhf(zib[2 * H2 + j] + r * zhb[2 * H2 + j]);
  float hn = (1.0f - z) * n + z * h[i];
  h[i] = hn;
  h_bf[i] = (bf16_t)hn;
  prebuf[((size_t)b * TT + t) * H2 + j] = hn;  // slot t never read at step t (mask j<t)
}

// ---------------------------------------------------------------------------
// Temporal encoder attention: e=q.enc_out, ee=exp(e), temporal normalization by
// running acc, softmax, context. One block per batch row.
// ---------------------------------------------------------------------------
__global__ __launch_bounds__(256) void s2s_att_enc(
    float* __restrict__ enc_att, float* __restrict__ enc_ctx, float* __restrict__ accb,
    const float* __restrict__ q, const float* __restrict__ enc_out, int t) {
  const int b = blockIdx.x, tid = threadIdx.x;
  __shared__ float sE[LL];
  __shared__ float sred[256];
  const float* qb = q + (size_t)b * H2;

  for (int l = tid; l < LL; l += 256) {
    const float* eo = enc_out + ((size_t)b * LL + l) * H2;
    float d = 0.0f;
    for (int k = 0; k < H2; ++k) d += qb[k] * eo[k];
    float ee = __expf(d);
    float a = accb[b * LL + l];
    sE[l] = (t == 0) ? ee : ee / (a + 1e-10f);
    accb[b * LL + l] = a + ee;
  }
  __syncthreads();

  float s = 0.0f;
  for (int l = tid; l < LL; l += 256) s += sE[l];
  sred[tid] = s;
  __syncthreads();
  for (int off = 128; off > 0; off >>= 1) {
    if (tid < off) sred[tid] += sred[tid + off];
    __syncthreads();
  }
  float inv = 1.0f / sred[0];

  for (int l = tid; l < LL; l += 256) {
    float a = sE[l] * inv;
    sE[l] = a;
    enc_att[b * LL + l] = a;
  }
  __syncthreads();

  for (int k = tid; k < H2; k += 256) {
    float accv = 0.0f;
    for (int l = 0; l < LL; ++l)
      accv += sE[l] * enc_out[((size_t)b * LL + l) * H2 + k];
    enc_ctx[(size_t)b * H2 + k] = accv;
  }
}

// Intra-decoder attention over prebuf rows j < t.
__global__ __launch_bounds__(128) void s2s_att_dec(
    float* __restrict__ dec_ctx, const float* __restrict__ q,
    const float* __restrict__ prebuf, int t) {
  const int b = blockIdx.x, tid = threadIdx.x;
  if (t == 0) {
    for (int k = tid; k < H2; k += 128) dec_ctx[(size_t)b * H2 + k] = 0.0f;
    return;
  }
  __shared__ float sE[TT];
  __shared__ float sred[128];
  const float* qb = q + (size_t)b * H2;

  for (int j = tid; j < t; j += 128) {
    const float* pv = prebuf + ((size_t)b * TT + j) * H2;
    float d = 0.0f;
    for (int k = 0; k < H2; ++k) d += qb[k] * pv[k];
    sE[j] = d;
  }
  __syncthreads();

  float mx = -1e30f;
  for (int j = tid; j < t; j += 128) mx = fmaxf(mx, sE[j]);
  sred[tid] = mx;
  __syncthreads();
  for (int off = 64; off > 0; off >>= 1) {
    if (tid < off) sred[tid] = fmaxf(sred[tid], sred[tid + off]);
    __syncthreads();
  }
  mx = sred[0];
  __syncthreads();

  float s = 0.0f;
  for (int j = tid; j < t; j += 128) s += __expf(sE[j] - mx);
  sred[tid] = s;
  __syncthreads();
  for (int off = 64; off > 0; off >>= 1) {
    if (tid < off) sred[tid] += sred[tid + off];
    __syncthreads();
  }
  float inv = 1.0f / sred[0];

  for (int j = tid; j < t; j += 128) sE[j] = __expf(sE[j] - mx) * inv;
  __syncthreads();

  for (int k = tid; k < H2; k += 128) {
    float accv = 0.0f;
    for (int j = 0; j < t; ++j)
      accv += sE[j] * prebuf[((size_t)b * TT + j) * H2 + k];
    dec_ctx[(size_t)b * H2 + k] = accv;
  }
}

// Build cat = [dec_h, enc_ctx, dec_ctx] (bf16 for WMMA) and p_gen = sigmoid(cat.Wp + bp)
__global__ __launch_bounds__(256) void s2s_cat_pgen(
    bf16_t* __restrict__ cat_bf, float* __restrict__ p_gen,
    const float* __restrict__ dec_h, const float* __restrict__ enc_ctx,
    const float* __restrict__ dec_ctx, const float* __restrict__ Wp,
    const float* __restrict__ bp) {
  const int b = blockIdx.x, tid = threadIdx.x;
  __shared__ float sred[256];
  float part = 0.0f;
  for (int i = tid; i < 3 * H2; i += 256) {
    float v = (i < H2) ? dec_h[(size_t)b * H2 + i]
              : (i < 2 * H2) ? enc_ctx[(size_t)b * H2 + i - H2]
                             : dec_ctx[(size_t)b * H2 + i - 2 * H2];
    cat_bf[(size_t)b * 3 * H2 + i] = (bf16_t)v;
    part += v * Wp[i];
  }
  sred[tid] = part;
  __syncthreads();
  for (int off = 128; off > 0; off >>= 1) {
    if (tid < off) sred[tid] += sred[tid + off];
    __syncthreads();
  }
  if (tid == 0) p_gen[b] = sigf(sred[0] + bp[0]);
}

// ---------------------------------------------------------------------------
// Final: softmax over vocab logits, mix with pointer distribution (scatter-add),
// argmax + loss + next input token. One block per batch row.
// ---------------------------------------------------------------------------
__global__ __launch_bounds__(256) void s2s_final_step(
    float* __restrict__ out, int* __restrict__ dec_in, float* __restrict__ finalb,
    const float* __restrict__ logits, const float* __restrict__ p_gen,
    const float* __restrict__ enc_att, const int* __restrict__ extend_vocab,
    const int* __restrict__ target_y, int t) {
  const int b = blockIdx.x, tid = threadIdx.x;
  __shared__ float sred[256];
  __shared__ int sidx[256];
  const float* lg = logits + (size_t)b * VV;
  float* fb = finalb + (size_t)b * VEXT;

  float mx = -1e30f;
  for (int j = tid; j < VV; j += 256) mx = fmaxf(mx, lg[j]);
  sred[tid] = mx;
  __syncthreads();
  for (int off = 128; off > 0; off >>= 1) {
    if (tid < off) sred[tid] = fmaxf(sred[tid], sred[tid + off]);
    __syncthreads();
  }
  mx = sred[0];
  __syncthreads();

  float s = 0.0f;
  for (int j = tid; j < VV; j += 256) s += __expf(lg[j] - mx);
  sred[tid] = s;
  __syncthreads();
  for (int off = 128; off > 0; off >>= 1) {
    if (tid < off) sred[tid] += sred[tid + off];
    __syncthreads();
  }
  float tot = sred[0];
  float pg = p_gen[b];
  float scale = (1.0f - pg) / tot;

  for (int j = tid; j < VEXT; j += 256)
    fb[j] = (j < VV) ? scale * __expf(lg[j] - mx) : 0.0f;
  __threadfence();
  __syncthreads();

  for (int l = tid; l < LL; l += 256)
    atomicAdd(&fb[extend_vocab[b * LL + l]], pg * enc_att[b * LL + l]);
  __threadfence();
  __syncthreads();

  float bv = -1e30f;
  int bi = 0;
  for (int j = tid; j < VEXT; j += 256) {
    float v = fb[j];
    if (v > bv) { bv = v; bi = j; }
  }
  sred[tid] = bv;
  sidx[tid] = bi;
  __syncthreads();
  for (int off = 128; off > 0; off >>= 1) {
    if (tid < off) {
      if (sred[tid + off] > sred[tid] ||
          (sred[tid + off] == sred[tid] && sidx[tid + off] < sidx[tid])) {
        sred[tid] = sred[tid + off];
        sidx[tid] = sidx[tid + off];
      }
    }
    __syncthreads();
  }
  if (tid == 0) {
    int sel = sidx[0];
    int tgt = target_y[b * TT + t];
    float p = fb[tgt];
    float loss = (tgt != 0) ? -__logf(p + 1e-12f) : 0.0f;
    out[b * TT + t] = (float)sel;             // ys
    out[BB * TT + b * TT + t] = loss;         // losses
    dec_in[b] = (sel >= VV) ? 1 : sel;        // UNK = 1
  }
}

// ---------------------------------------------------------------------------
// Host orchestration
// ---------------------------------------------------------------------------
static inline size_t alignup(size_t x) { return (x + 255) & ~(size_t)255; }

extern "C" void kernel_launch(void* const* d_in, const int* in_sizes, int n_in,
                              void* d_out, int out_size, void* d_ws, size_t ws_size,
                              hipStream_t stream) {
  (void)in_sizes; (void)n_in; (void)out_size; (void)ws_size;
  const int*   x          = (const int*)d_in[0];
  const int*   target_y   = (const int*)d_in[2];
  const int*   ext_vocab  = (const int*)d_in[3];
  const float* embedding  = (const float*)d_in[5];
  const float* enc_Wi_f   = (const float*)d_in[6];
  const float* enc_Wh_f   = (const float*)d_in[7];
  const float* enc_b_f    = (const float*)d_in[8];
  const float* enc_Wi_b   = (const float*)d_in[9];
  const float* enc_Wh_b   = (const float*)d_in[10];
  const float* enc_b_b    = (const float*)d_in[11];
  const float* dec_Wi     = (const float*)d_in[12];
  const float* dec_Wh     = (const float*)d_in[13];
  const float* dec_b      = (const float*)d_in[14];
  const float* W_enc_att  = (const float*)d_in[15];
  const float* W_dec_att  = (const float*)d_in[16];
  const float* Wv         = (const float*)d_in[17];
  const float* bv         = (const float*)d_in[18];
  const float* Wp         = (const float*)d_in[19];
  const float* bp         = (const float*)d_in[20];
  float* out = (float*)d_out;

  // Workspace carving
  char* p = (char*)d_ws;
  auto alloc = [&](size_t bytes) -> char* { char* r = p; p += alignup(bytes); return r; };

  bf16_t* xe      = (bf16_t*)alloc((size_t)LL * BB * EE * 2);
  float*  pre_f   = (float*) alloc((size_t)LL * BB * 4 * HH * 4);
  float*  pre_b   = (float*) alloc((size_t)LL * BB * 4 * HH * 4);
  float*  enc_out = (float*) alloc((size_t)BB * LL * H2 * 4);

  bf16_t* Wi_f_t  = (bf16_t*)alloc((size_t)(4 * HH) * EE * 2);
  bf16_t* Wh_f_t  = (bf16_t*)alloc((size_t)(4 * HH) * HH * 2);
  bf16_t* Wi_b_t  = (bf16_t*)alloc((size_t)(4 * HH) * EE * 2);
  bf16_t* Wh_b_t  = (bf16_t*)alloc((size_t)(4 * HH) * HH * 2);
  bf16_t* dWi_t   = (bf16_t*)alloc((size_t)(3 * H2) * EE * 2);
  bf16_t* dWh_t   = (bf16_t*)alloc((size_t)(3 * H2) * H2 * 2);
  bf16_t* Wea_t   = (bf16_t*)alloc((size_t)H2 * H2 * 2);
  bf16_t* Wda_t   = (bf16_t*)alloc((size_t)H2 * H2 * 2);
  bf16_t* Wv_t    = (bf16_t*)alloc((size_t)VV * (3 * H2) * 2);

  int*    dec_in   = (int*)   alloc(BB * 4);
  float*  dec_h    = (float*) alloc((size_t)BB * H2 * 4);
  bf16_t* dec_h_bf = (bf16_t*)alloc((size_t)BB * H2 * 2);
  bf16_t* emb_bf   = (bf16_t*)alloc((size_t)BB * EE * 2);
  float*  zi       = (float*) alloc((size_t)BB * 3 * H2 * 4);
  float*  zh       = (float*) alloc((size_t)BB * 3 * H2 * 4);
  float*  q_enc    = (float*) alloc((size_t)BB * H2 * 4);
  float*  q_dec    = (float*) alloc((size_t)BB * H2 * 4);
  float*  accb     = (float*) alloc((size_t)BB * LL * 4);
  float*  enc_att  = (float*) alloc((size_t)BB * LL * 4);
  float*  enc_ctx  = (float*) alloc((size_t)BB * H2 * 4);
  float*  dec_ctx  = (float*) alloc((size_t)BB * H2 * 4);
  float*  prebuf   = (float*) alloc((size_t)BB * TT * H2 * 4);
  bf16_t* cat_bf   = (bf16_t*)alloc((size_t)BB * 3 * H2 * 2);
  float*  p_gen    = (float*) alloc((size_t)BB * 4);
  float*  logits   = (float*) alloc((size_t)BB * VV * 4);
  float*  finalb   = (float*) alloc((size_t)BB * VEXT * 4);

  auto ltrans = [&](bf16_t* Wt, const float* W, int K, int N) {
    size_t tot = (size_t)K * N;
    s2s_transpose_bf16<<<dim3((unsigned)((tot + 255) / 256)), 256, 0, stream>>>(Wt, W, K, N);
  };

  // --- One-time (per call) weight conversion to bf16, transposed [N][K] ---
  ltrans(Wi_f_t, enc_Wi_f, EE, 4 * HH);
  ltrans(Wh_f_t, enc_Wh_f, HH, 4 * HH);
  ltrans(Wi_b_t, enc_Wi_b, EE, 4 * HH);
  ltrans(Wh_b_t, enc_Wh_b, HH, 4 * HH);
  ltrans(dWi_t, dec_Wi, EE, 3 * H2);
  ltrans(dWh_t, dec_Wh, H2, 3 * H2);
  ltrans(Wea_t, W_enc_att, H2, H2);
  ltrans(Wda_t, W_dec_att, H2, H2);
  ltrans(Wv_t, Wv, 3 * H2, VV);

  // --- Encoder ---
  {
    size_t tot = (size_t)LL * BB * EE;
    s2s_embed_enc<<<dim3((unsigned)((tot + 255) / 256)), 256, 0, stream>>>(xe, x, embedding);
  }
  // pre = xe @ Wi + b  (M = L*B = 12800, fully parallel; hoists Wi out of seq loop)
  s2s_gemm_wmma<<<dim3(LL * BB / 32, 4 * HH / 32), 32, 0, stream>>>(
      pre_f, xe, Wi_f_t, enc_b_f, LL * BB, 4 * HH, EE);
  s2s_gemm_wmma<<<dim3(LL * BB / 32, 4 * HH / 32), 32, 0, stream>>>(
      pre_b, xe, Wi_b_t, enc_b_b, LL * BB, 4 * HH, EE);
  // sequential h@Wh recurrence (both directions in parallel)
  s2s_lstm_seq<<<2, 512, 0, stream>>>(enc_out, pre_f, pre_b, Wh_f_t, Wh_b_t);

  // --- Decoder ---
  s2s_dec_init<<<(BB * LL + 255) / 256, 256, 0, stream>>>(dec_in, dec_h, dec_h_bf, accb, enc_out);

  for (int t = 0; t < TT; ++t) {
    s2s_embed_dec<<<(BB * EE + 255) / 256, 256, 0, stream>>>(emb_bf, dec_in, embedding);
    s2s_gemm_wmma<<<dim3(BB / 32, 3 * H2 / 32), 32, 0, stream>>>(
        zi, emb_bf, dWi_t, dec_b, BB, 3 * H2, EE);
    s2s_gemm_wmma<<<dim3(BB / 32, 3 * H2 / 32), 32, 0, stream>>>(
        zh, dec_h_bf, dWh_t, nullptr, BB, 3 * H2, H2);
    s2s_gru_combine<<<(BB * H2 + 255) / 256, 256, 0, stream>>>(dec_h, dec_h_bf, prebuf, zi, zh, t);
    s2s_gemm_wmma<<<dim3(BB / 32, H2 / 32), 32, 0, stream>>>(
        q_enc, dec_h_bf, Wea_t, nullptr, BB, H2, H2);
    s2s_gemm_wmma<<<dim3(BB / 32, H2 / 32), 32, 0, stream>>>(
        q_dec, dec_h_bf, Wda_t, nullptr, BB, H2, H2);
    s2s_att_enc<<<BB, 256, 0, stream>>>(enc_att, enc_ctx, accb, q_enc, enc_out, t);
    s2s_att_dec<<<BB, 128, 0, stream>>>(dec_ctx, q_dec, prebuf, t);
    s2s_cat_pgen<<<BB, 256, 0, stream>>>(cat_bf, p_gen, dec_h, enc_ctx, dec_ctx, Wp, bp);
    // The big one: 32x32000x3072 bf16 WMMA GEMM; Wv read exactly once per step
    s2s_gemm_wmma<<<dim3(BB / 32, VV / 32), 32, 0, stream>>>(
        logits, cat_bf, Wv_t, bv, BB, VV, 3 * H2);
    s2s_final_step<<<BB, 256, 0, stream>>>(out, dec_in, finalb, logits, p_gen, enc_att,
                                           ext_vocab, target_y, t);
  }
}
